// EchoAgentAttention_19713899889504
// MI455X (gfx1250) — matrix operand order repaired and checked by reading, not compiled, not run on hardware
//
#include <hip/hip_runtime.h>

// ---------- problem constants (fixed by the harness/reference) ----------
#define BB     16          // B
#define TT     10          // T
#define BT     160         // B*T
#define NTOK   257
#define CC     768
#define HEADS  8
#define HD     96
#define AGENTS 32
#define S3     2304        // 3*C
#define M_ALL  41120       // BT*NTOK
#define M_AG   5120        // BB * TT*AGENTS
#define SMHA   320         // TT*AGENTS
#define SCALE_HD 0.10206207261596575f   // 96^-0.5

// ---------- bf16 helpers (manual RNE, storage = unsigned short) ----------
__device__ __forceinline__ unsigned short f2bf(float f) {
  unsigned int u = __float_as_uint(f);
  u += 0x7FFFu + ((u >> 16) & 1u);
  return (unsigned short)(u >> 16);
}
__device__ __forceinline__ float bf2f(unsigned short s) {
  return __uint_as_float(((unsigned int)s) << 16);
}

// ---------- WMMA types ----------
typedef __attribute__((ext_vector_type(16))) __bf16 v16bf;
typedef __attribute__((ext_vector_type(8)))  __bf16 v8bf;
typedef __attribute__((ext_vector_type(8)))  float  v8f;

// =====================================================================
// bf16 GEMM:  C[M,N] = A[M,K] * B[N,K]^T (+bias[N]) (+addm[M,N])
// A,B bf16 row-major (ld=K). Output fp32 or bf16 (compile-time MODE).
// MODE bit0: add bias[N]; bit1: add addm[M,N]; bit2: store bf16.
// Block = 256 thr = 8 waves. Wave tile = 32M x 128N (16 wmma / K-step).
// Block tile = 128M x 256N. B tile staged in LDS (double-buffered) via
// gfx1250 global_load_async_to_lds_b128 + s_wait_asynccnt.
// Requires: N % 256 == 0, K % 32 == 0, M % 32 == 0 or guarded wave rows.
// =====================================================================
template<int MODE>
__global__ void __launch_bounds__(256) k_gemm_bf16(
    const unsigned short* __restrict__ A, const unsigned short* __restrict__ Bw,
    void* __restrict__ Cout, int M, int N, int K,
    const float* __restrict__ bias, const float* __restrict__ addm)
{
  __shared__ alignas(64) unsigned short bufB[2][256 * 32];  // 2 x 16KB

  const int lane = threadIdx.x & 31;
  const int wave = threadIdx.x >> 5;
  const int wm = wave >> 1;                 // 0..3  (32-row M group)
  const int wn = wave & 1;                  // 0..1  (128-col N group)
  const int tileM = (blockIdx.x * 4 + wm) * 32;
  const int nBlk  = blockIdx.y * 256;
  const bool active = tileM < M;

  const int l15  = lane & 15;
  const int half = lane >> 4;               // K-half select
  const int kbA  = half * 16;               // A lane K sub-base (elements)
  const int kbB  = half * 16;               // B lane K sub-base (elements)

  v8f acc[2][8];
  #pragma unroll
  for (int mi = 0; mi < 2; ++mi)
    #pragma unroll
    for (int j = 0; j < 8; ++j)
      #pragma unroll
      for (int e = 0; e < 8; ++e) acc[mi][j][e] = 0.f;

  const unsigned short* aBase = A + (size_t)(tileM + l15) * K;
  const unsigned short* bBlk  = Bw + (size_t)nBlk * K;

  // async-stage one 256N x 32K bf16 tile into bufB[buf] (whole block helps)
  auto stage = [&](int buf, int k0) {
    #pragma unroll
    for (int i = 0; i < 4; ++i) {
      const int c   = (int)threadIdx.x + 256 * i;   // 1024 x 16B chunks
      const int n   = c >> 2;
      const int sub = c & 3;
      unsigned long long ga =
          (unsigned long long)(size_t)(bBlk + (size_t)n * K + k0 + sub * 8);
      unsigned lo = (unsigned)(size_t)(&bufB[buf][n * 32 + sub * 8]);
      asm volatile("global_load_async_to_lds_b128 %0, %1, off"
                   :: "v"(lo), "v"(ga) : "memory");
    }
  };

  stage(0, 0);                               // prologue
  int cur = 0;
  for (int k0 = 0; k0 < K; k0 += 32) {
    asm volatile("s_wait_asynccnt 0x0" ::: "memory");
    __syncthreads();                         // bufB[cur] ready, prev reads done
    if (k0 + 32 < K) stage(cur ^ 1, k0 + 32);

    if (active) {
      __builtin_prefetch(aBase + k0 + 64, 0, 3);       // near-cache prefetch
      v16bf afrag[2];
      #pragma unroll
      for (int mi = 0; mi < 2; ++mi) {
        const unsigned short* ar = aBase + (size_t)(16 * mi) * K + k0 + kbA;
        v8bf alo = *reinterpret_cast<const v8bf*>(ar);
        v8bf ahi = *reinterpret_cast<const v8bf*>(ar + 8);
        afrag[mi] = __builtin_shufflevector(alo, ahi,
            0,1,2,3,4,5,6,7,8,9,10,11,12,13,14,15);
      }
      #pragma unroll
      for (int j = 0; j < 8; ++j) {
        const unsigned short* bp =
            &bufB[cur][(wn * 128 + j * 16 + l15) * 32 + kbB];
        v16bf bfrag = *reinterpret_cast<const v16bf*>(bp);   // ds_load
        acc[0][j] = __builtin_amdgcn_wmma_f32_16x16x32_bf16(
            false, afrag[0], false, bfrag, (short)0, acc[0][j], false, false);
        acc[1][j] = __builtin_amdgcn_wmma_f32_16x16x32_bf16(
            false, afrag[1], false, bfrag, (short)0, acc[1][j], false, false);
      }
    }
    cur ^= 1;
  }

  if (!active) return;
  // Epilogue. D layout: lane l -> n=l&15, vgpr v -> m = v + 8*(l>>4)
  #pragma unroll
  for (int mi = 0; mi < 2; ++mi) {
    #pragma unroll
    for (int j = 0; j < 8; ++j) {
      const int col = nBlk + wn * 128 + j * 16 + l15;
      const float bv = (MODE & 1) ? bias[col] : 0.f;   // one load per column
      #pragma unroll
      for (int v = 0; v < 8; ++v) {
        const int row = tileM + mi * 16 + v + half * 8;
        float val = acc[mi][j][v] + bv;
        const size_t o = (size_t)row * N + col;
        if (MODE & 2) val += addm[o];
        if (MODE & 4) ((unsigned short*)Cout)[o] = f2bf(val);
        else          ((float*)Cout)[o] = val;
      }
    }
  }
}

// ---------- fp32 -> bf16 convert ----------
__global__ void __launch_bounds__(256) k_cvt(
    const float* __restrict__ s, unsigned short* __restrict__ d, int n)
{
  int i = blockIdx.x * 256 + threadIdx.x;
  if (i < n) d[i] = f2bf(s[i]);
}

// ---------- x + pos_embed -> bf16 ----------
__global__ void __launch_bounds__(256) k_addpos(
    const float* __restrict__ x, const float* __restrict__ pos,
    unsigned short* __restrict__ d, int n)
{
  int i = blockIdx.x * 256 + threadIdx.x;
  if (i < n) d[i] = f2bf(x[i] + pos[i % (NTOK * CC)]);
}

// ---------- affinity aff[b,i] = <imgn_i, clsn> (on x+pos) ----------
__global__ void __launch_bounds__(128) k_aff(
    const float* __restrict__ x, const float* __restrict__ pos,
    float* __restrict__ aff)
{
  const int blk = blockIdx.x;                    // b*256 + i
  const int b = blk >> 8, i = blk & 255;
  const int tid = threadIdx.x;
  __shared__ float rd[128], rn[128], rc[128];
  const float* xi = x + ((size_t)(b * NTOK + 1 + i)) * CC;
  const float* xc = x + ((size_t)(b * NTOK)) * CC;
  const float* pi = pos + (size_t)(1 + i) * CC;
  float d = 0.f, ni = 0.f, nc = 0.f;
  for (int e = tid; e < CC; e += 128) {
    float a = xi[e] + pi[e];
    float c = xc[e] + pos[e];
    d += a * c; ni += a * a; nc += c * c;
  }
  rd[tid] = d; rn[tid] = ni; rc[tid] = nc; __syncthreads();
  for (int s = 64; s > 0; s >>= 1) {
    if (tid < s) { rd[tid] += rd[tid+s]; rn[tid] += rn[tid+s]; rc[tid] += rc[tid+s]; }
    __syncthreads();
  }
  if (tid == 0) {
    float den = fmaxf(sqrtf(rn[0]), 1e-12f) * fmaxf(sqrtf(rc[0]), 1e-12f);
    aff[b * 256 + i] = rd[0] / den;
  }
}

// ---------- top-32 per batch (descending, lower index wins ties) ----------
__global__ void __launch_bounds__(256) k_topk(
    const float* __restrict__ aff, int* __restrict__ idx)
{
  const int b = blockIdx.x;
  const int tid = threadIdx.x;
  __shared__ float vals[256];
  __shared__ float rv[256];
  __shared__ int   ri[256];
  vals[tid] = aff[b * 256 + tid];
  __syncthreads();
  for (int it = 0; it < AGENTS; ++it) {
    rv[tid] = vals[tid]; ri[tid] = tid;
    __syncthreads();
    for (int s = 128; s > 0; s >>= 1) {
      if (tid < s) {
        if (rv[tid+s] > rv[tid] || (rv[tid+s] == rv[tid] && ri[tid+s] < ri[tid])) {
          rv[tid] = rv[tid+s]; ri[tid] = ri[tid+s];
        }
      }
      __syncthreads();
    }
    if (tid == 0) { idx[b * AGENTS + it] = ri[0]; vals[ri[0]] = -3.0e38f; }
    __syncthreads();
  }
}

// ---------- gather agent rows from q + tem_embed -> bf16 ----------
__global__ void __launch_bounds__(256) k_gather(
    const unsigned short* __restrict__ qkvbf, const int* __restrict__ idx,
    const float* __restrict__ tem, unsigned short* __restrict__ agin)
{
  int i = blockIdx.x * 256 + threadIdx.x;
  if (i >= M_AG * CC) return;
  int r = i / CC, c = i % CC;
  int bt = r >> 5, a = r & 31, t = bt % TT;
  int tok = idx[bt * AGENTS + a] + 1;
  float v = bf2f(qkvbf[((size_t)(bt * NTOK + tok)) * S3 + c]) + tem[t * CC + c];
  agin[i] = f2bf(v);
}

// ---------- inner MHA attention over S=320 (fp32 qkv2) -> bf16 out ----------
__global__ void __launch_bounds__(128) k_mha_attn(
    const float* __restrict__ qkv2, unsigned short* __restrict__ mo)
{
  const int blk = blockIdx.x;                 // b*(8*320) + h*320 + s
  const int b = blk / (HEADS * SMHA);
  const int rem = blk % (HEADS * SMHA);
  const int h = rem / SMHA;
  const int s = rem % SMHA;
  const int tid = threadIdx.x;
  __shared__ float qv[HD];
  __shared__ float sc[SMHA];
  __shared__ float red[128];
  if (tid < HD)
    qv[tid] = qkv2[((size_t)(b * SMHA + s)) * S3 + h * HD + tid] * SCALE_HD;
  __syncthreads();
  float lmax = -1e30f;
  for (int t = tid; t < SMHA; t += 128) {
    const float* krow = qkv2 + ((size_t)(b * SMHA + t)) * S3 + CC + h * HD;
    float d = 0.f;
    for (int e = 0; e < HD; ++e) d += qv[e] * krow[e];
    sc[t] = d;
    lmax = fmaxf(lmax, d);
  }
  red[tid] = lmax; __syncthreads();
  for (int st = 64; st > 0; st >>= 1) {
    if (tid < st) red[tid] = fmaxf(red[tid], red[tid + st]);
    __syncthreads();
  }
  const float m = red[0];
  __syncthreads();
  float lsum = 0.f;
  for (int t = tid; t < SMHA; t += 128) {
    float p = __expf(sc[t] - m); sc[t] = p; lsum += p;
  }
  red[tid] = lsum; __syncthreads();
  for (int st = 64; st > 0; st >>= 1) {
    if (tid < st) red[tid] += red[tid + st];
    __syncthreads();
  }
  const float inv = 1.f / red[0];
  __syncthreads();
  if (tid < HD) {
    float acc = 0.f;
    for (int t = 0; t < SMHA; ++t)
      acc += sc[t] * qkv2[((size_t)(b * SMHA + t)) * S3 + 2 * CC + h * HD + tid];
    mo[((size_t)(b * SMHA + s)) * CC + h * HD + tid] = f2bf(acc * inv);
  }
}

// ---------- attn1 + agent_v: per (b,t,h) block ----------
__global__ void __launch_bounds__(256) k_attn1(
    const float* __restrict__ agent, const unsigned short* __restrict__ qkvbf,
    float* __restrict__ agv)
{
  const int blk = blockIdx.x;                  // b*80 + t*8 + h
  const int b = blk / (TT * HEADS);
  const int th = blk % (TT * HEADS);
  const int t = th / HEADS, h = th % HEADS;
  const int bt = b * TT + t;
  const int tid = threadIdx.x;
  __shared__ float ag[AGENTS * HD];            // 12 KB
  __shared__ float sc[AGENTS * NTOK];          // 32.9 KB (big WGP LDS)
  for (int i = tid; i < AGENTS * HD; i += 256) {
    int a = i / HD, d = i % HD;
    ag[i] = agent[((size_t)(bt * AGENTS + a)) * CC + h * HD + d] * SCALE_HD;
  }
  __syncthreads();
  for (int n = tid; n < NTOK; n += 256) {
    const unsigned short* krow = qkvbf + ((size_t)(bt * NTOK + n)) * S3 + CC + h * HD;
    float acc[AGENTS];
    #pragma unroll
    for (int a = 0; a < AGENTS; ++a) acc[a] = 0.f;
    for (int d = 0; d < HD; ++d) {
      float kv = bf2f(krow[d]);
      #pragma unroll
      for (int a = 0; a < AGENTS; ++a) acc[a] += ag[a * HD + d] * kv;
    }
    #pragma unroll
    for (int a = 0; a < AGENTS; ++a) sc[a * NTOK + n] = acc[a];
  }
  __syncthreads();
  if (tid < AGENTS) {
    float* row = sc + tid * NTOK;
    float m = -1e30f;
    for (int n = 0; n < NTOK; ++n) m = fmaxf(m, row[n]);
    float ssum = 0.f;
    for (int n = 0; n < NTOK; ++n) { float p = __expf(row[n] - m); row[n] = p; ssum += p; }
    float inv = 1.f / ssum;
    for (int n = 0; n < NTOK; ++n) row[n] *= inv;
  }
  __syncthreads();
  for (int o = tid; o < AGENTS * HD; o += 256) {
    int a = o / HD, d = o % HD;
    const float* prow = sc + a * NTOK;
    float acc = 0.f;
    for (int n = 0; n < NTOK; ++n)
      acc += prow[n] * bf2f(qkvbf[((size_t)(bt * NTOK + n)) * S3 + 2 * CC + h * HD + d]);
    agv[(size_t)blk * (AGENTS * HD) + o] = acc;
  }
}

// ---------- attn2 + fx: per (b,t,h) block ----------
__global__ void __launch_bounds__(256) k_attn2fx(
    const float* __restrict__ agent, const float* __restrict__ agv,
    const unsigned short* __restrict__ qkvbf, float* __restrict__ xo)
{
  const int blk = blockIdx.x;
  const int b = blk / (TT * HEADS);
  const int th = blk % (TT * HEADS);
  const int t = th / HEADS, h = th % HEADS;
  const int bt = b * TT + t;
  const int tid = threadIdx.x;
  __shared__ float ag[AGENTS * HD];
  __shared__ float av[AGENTS * HD];
  for (int i = tid; i < AGENTS * HD; i += 256) {
    int a = i / HD, d = i % HD;
    ag[i] = agent[((size_t)(bt * AGENTS + a)) * CC + h * HD + d] * SCALE_HD;
    av[i] = agv[(size_t)blk * (AGENTS * HD) + i];
  }
  __syncthreads();
  for (int n = tid; n < NTOK; n += 256) {
    const unsigned short* qrow = qkvbf + ((size_t)(bt * NTOK + n)) * S3 + h * HD;
    float e[AGENTS];
    #pragma unroll
    for (int a = 0; a < AGENTS; ++a) e[a] = 0.f;
    for (int d = 0; d < HD; ++d) {
      float qv = bf2f(qrow[d]);
      #pragma unroll
      for (int a = 0; a < AGENTS; ++a) e[a] += qv * ag[a * HD + d];
    }
    float m = -1e30f;
    #pragma unroll
    for (int a = 0; a < AGENTS; ++a) m = fmaxf(m, e[a]);
    float ssum = 0.f;
    #pragma unroll
    for (int a = 0; a < AGENTS; ++a) { e[a] = __expf(e[a] - m); ssum += e[a]; }
    const float inv = 1.f / ssum;
    float* orow = xo + ((size_t)(bt * NTOK + n)) * CC + h * HD;
    for (int d = 0; d < HD; ++d) {
      float acc = 0.f;
      #pragma unroll
      for (int a = 0; a < AGENTS; ++a) acc += e[a] * av[a * HD + d];
      orow[d] = acc * inv;
    }
  }
}

// ---------- depthwise 3x3 conv on v-image, add into xo tokens 1..256 ----------
__global__ void __launch_bounds__(256) k_dwc(
    const unsigned short* __restrict__ qkvbf, const float* __restrict__ w,
    const float* __restrict__ bias, float* __restrict__ xo)
{
  const int blk = blockIdx.x;                 // bt*256 + pixel
  const int bt = blk >> 8;
  const int p = blk & 255;
  const int y = p >> 4, xg = p & 15;
  for (int c = threadIdx.x; c < CC; c += 256) {
    float acc = bias[c];
    #pragma unroll
    for (int ky = -1; ky <= 1; ++ky) {
      int yy = y + ky; if (yy < 0 || yy > 15) continue;
      #pragma unroll
      for (int kx = -1; kx <= 1; ++kx) {
        int xx = xg + kx; if (xx < 0 || xx > 15) continue;
        int tok = 1 + yy * 16 + xx;
        float v = bf2f(qkvbf[((size_t)(bt * NTOK + tok)) * S3 + 2 * CC + c]);
        acc += v * w[((ky + 1) * 3 + (kx + 1)) * CC + c];
      }
    }
    xo[((size_t)(bt * NTOK + 1 + p)) * CC + c] += acc;
  }
}

// ---------- layer norm per row of 768; output fp32 or bf16 ----------
__global__ void __launch_bounds__(256) k_ln(
    const float* __restrict__ in, const float* __restrict__ g,
    const float* __restrict__ b, float* __restrict__ outF,
    unsigned short* __restrict__ outBf)
{
  const int r = blockIdx.x;
  const int tid = threadIdx.x;
  __shared__ float red[256];
  const float* row = in + (size_t)r * CC;
  float s = 0.f;
  for (int e = tid; e < CC; e += 256) s += row[e];
  red[tid] = s; __syncthreads();
  for (int st = 128; st > 0; st >>= 1) { if (tid < st) red[tid] += red[tid+st]; __syncthreads(); }
  const float mean = red[0] / (float)CC;
  __syncthreads();
  float v = 0.f;
  for (int e = tid; e < CC; e += 256) { float d = row[e] - mean; v += d * d; }
  red[tid] = v; __syncthreads();
  for (int st = 128; st > 0; st >>= 1) { if (tid < st) red[tid] += red[tid+st]; __syncthreads(); }
  const float rstd = rsqrtf(red[0] / (float)CC + 1e-5f);
  for (int e = tid; e < CC; e += 256) {
    float o = (row[e] - mean) * rstd * g[e] + b[e];
    if (outBf) outBf[(size_t)r * CC + e] = f2bf(o);
    else       outF [(size_t)r * CC + e] = o;
  }
}

// =====================================================================
extern "C" void kernel_launch(void* const* d_in, const int* in_sizes, int n_in,
                              void* d_out, int out_size, void* d_ws, size_t ws_size,
                              hipStream_t stream)
{
  const float* x      = (const float*)d_in[0];
  const float* pos    = (const float*)d_in[2];
  const float* tem    = (const float*)d_in[3];
  const float* qkv_w  = (const float*)d_in[4];
  const float* proj_w = (const float*)d_in[5];
  const float* proj_b = (const float*)d_in[6];
  const float* min_w  = (const float*)d_in[7];
  const float* min_b  = (const float*)d_in[8];
  const float* mout_w = (const float*)d_in[9];
  const float* mout_b = (const float*)d_in[10];
  const float* dwcw   = (const float*)d_in[11];
  const float* dwcb   = (const float*)d_in[12];
  const float* ln1g   = (const float*)d_in[13];
  const float* ln1b   = (const float*)d_in[14];
  const float* ln2g   = (const float*)d_in[15];
  const float* ln2b   = (const float*)d_in[16];

  // ---- workspace layout (256B aligned, with aliasing) ----
  size_t off = 0;
  auto take = [&](size_t bytes) { size_t o = off; off = (off + bytes + 255) & ~(size_t)255; return o; };
  const size_t o_xpbf  = take((size_t)M_ALL * CC * 2);   // x+pos bf16 (later reused as ln1 bf16)
  const size_t o_wqkv  = take((size_t)S3 * CC * 2);
  const size_t o_wmin  = take((size_t)S3 * CC * 2);
  const size_t o_wmout = take((size_t)CC * CC * 2);
  const size_t o_wproj = take((size_t)CC * CC * 2);
  const size_t o_aff   = take((size_t)BT * 256 * 4);
  const size_t o_idx   = take((size_t)BT * AGENTS * 4);
  const size_t o_qkvbf = take((size_t)M_ALL * S3 * 2);   // qkv bf16 (later reused as xo2 fp32)
  const size_t o_agin  = take((size_t)M_AG * CC * 2);
  const size_t o_qkv2  = take((size_t)M_AG * S3 * 4);
  const size_t o_mobf  = take((size_t)M_AG * CC * 2);
  const size_t o_agent = take((size_t)M_AG * CC * 4);
  const size_t o_agv   = take((size_t)BT * HEADS * AGENTS * HD * 4);
  const size_t o_xo    = take((size_t)M_ALL * CC * 4);
  if (off > ws_size) return;                              // insufficient scratch
  char* ws = (char*)d_ws;
  unsigned short* xpbf  = (unsigned short*)(ws + o_xpbf);
  unsigned short* wqkv  = (unsigned short*)(ws + o_wqkv);
  unsigned short* wmin  = (unsigned short*)(ws + o_wmin);
  unsigned short* wmout = (unsigned short*)(ws + o_wmout);
  unsigned short* wproj = (unsigned short*)(ws + o_wproj);
  float*          aff   = (float*)(ws + o_aff);
  int*            idx   = (int*)(ws + o_idx);
  unsigned short* qkvbf = (unsigned short*)(ws + o_qkvbf);
  unsigned short* agin  = (unsigned short*)(ws + o_agin);
  float*          qkv2  = (float*)(ws + o_qkv2);
  unsigned short* mobf  = (unsigned short*)(ws + o_mobf);
  float*          agent = (float*)(ws + o_agent);
  float*          agv   = (float*)(ws + o_agv);
  float*          xo    = (float*)(ws + o_xo);
  unsigned short* ln1bf = xpbf;            // alias: xpbf dead after qkv GEMM
  float*          xo2   = (float*)qkvbf;   // alias: qkvbf dead after dwc/attn kernels

  // 1. weight conversions to bf16
  k_cvt<<<(S3*CC + 255)/256, 256, 0, stream>>>(qkv_w,  wqkv,  S3*CC);
  k_cvt<<<(S3*CC + 255)/256, 256, 0, stream>>>(min_w,  wmin,  S3*CC);
  k_cvt<<<(CC*CC + 255)/256, 256, 0, stream>>>(mout_w, wmout, CC*CC);
  k_cvt<<<(CC*CC + 255)/256, 256, 0, stream>>>(proj_w, wproj, CC*CC);
  // 2. x + pos -> bf16
  k_addpos<<<(M_ALL*CC + 255)/256, 256, 0, stream>>>(x, pos, xpbf, M_ALL*CC);
  // 3. affinity + top-k agents
  k_aff<<<BT*256, 128, 0, stream>>>(x, pos, aff);
  k_topk<<<BT, 256, 0, stream>>>(aff, idx);
  // 4. qkv = (x+pos) @ qkv_w^T  (bf16 out)   M=41120 -> 1285 32-row tiles
  k_gemm_bf16<4><<<dim3((M_ALL/32 + 3)/4, S3/256), 256, 0, stream>>>(
      xpbf, wqkv, (void*)qkvbf, M_ALL, S3, CC, nullptr, nullptr);
  // 5. gather agents (+tem_embed)
  k_gather<<<(M_AG*CC + 255)/256, 256, 0, stream>>>(qkvbf, idx, tem, agin);
  // 6. inner MHA: in-proj, attention, out-proj
  k_gemm_bf16<1><<<dim3((M_AG/32 + 3)/4, S3/256), 256, 0, stream>>>(
      agin, wmin, (void*)qkv2, M_AG, S3, CC, min_b, nullptr);
  k_mha_attn<<<BB*HEADS*SMHA, 128, 0, stream>>>(qkv2, mobf);
  k_gemm_bf16<1><<<dim3((M_AG/32 + 3)/4, CC/256), 256, 0, stream>>>(
      mobf, wmout, (void*)agent, M_AG, CC, CC, mout_b, nullptr);
  // 7. agent attention: attn1 -> agent_v, attn2 -> fx (=xo)
  k_attn1<<<BT*HEADS, 256, 0, stream>>>(agent, qkvbf, agv);
  k_attn2fx<<<BT*HEADS, 256, 0, stream>>>(agent, agv, qkvbf, xo);
  // 8. depthwise conv on v, add into xo[:,1:]
  k_dwc<<<BT*256, 256, 0, stream>>>(qkvbf, dwcw, dwcb, xo);
  // 9. LN1 -> bf16, proj GEMM (+bias +identity), LN2 -> out
  k_ln<<<M_ALL, 256, 0, stream>>>(xo, ln1g, ln1b, nullptr, ln1bf);
  k_gemm_bf16<3><<<dim3((M_ALL/32 + 3)/4, CC/256), 256, 0, stream>>>(
      ln1bf, wproj, (void*)xo2, M_ALL, CC, CC, proj_b, x);
  k_ln<<<M_ALL, 256, 0, stream>>>(xo2, ln2g, ln2b, (float*)d_out, nullptr);
}
